// MatchingNetwork_31026843746841
// MI455X (gfx1250) — matrix operand order repaired and checked by reading, not compiled, not run on hardware
//
#include <hip/hip_runtime.h>
#include <hip/hip_bf16.h>

typedef __attribute__((ext_vector_type(16))) _Float16 v16h;
typedef __attribute__((ext_vector_type(8)))  _Float16 v8h;
typedef __attribute__((ext_vector_type(8)))  float    v8f;
typedef int b128_t __attribute__((vector_size(16)));   // matches async-copy builtin param

#define N_NODES 32768
#define N_EDGES 1048576
#define NUM_GRAPHS 256
#define IN_DIM 28
#define EDGE_DIM 20
#define HID 3000
#define HIDP 3008
#define PROJ 3750
#define PROJP 3776
#define OUT_DIM 128
#define K1P 32
#define BN_EPS 1e-5f

#define LDS_STRIDE 40  // 32 halves + 8 pad -> 80B row pitch, conflict-free frag reads

#if __has_builtin(__builtin_amdgcn_global_load_async_to_lds_b128)
#define HAVE_ASYNC_CP 1
#else
#define HAVE_ASYNC_CP 0
#endif

#define AS_GLOBAL __attribute__((address_space(1)))
#define AS_LDS    __attribute__((address_space(3)))

// 16-byte global -> LDS copy (async on CDNA5, sync fallback otherwise)
__device__ __forceinline__ void cp16(const _Float16* g, _Float16* l) {
#if HAVE_ASYNC_CP
    __builtin_amdgcn_global_load_async_to_lds_b128(
        (AS_GLOBAL b128_t*)g, (AS_LDS b128_t*)l, 0, 0);
#else
    *(v8h*)l = *(const v8h*)g;
#endif
}

__device__ __forceinline__ void wait_cp() {
#if HAVE_ASYNC_CP
#if __has_builtin(__builtin_amdgcn_s_wait_asynccnt)
    __builtin_amdgcn_s_wait_asynccnt(0);
#else
    asm volatile("s_wait_asynccnt 0" ::: "memory");
#endif
#endif
}

// ---------------------------------------------------------------- utilities
__global__ void zero_f32(float* p, size_t n) {
    size_t i = (size_t)blockIdx.x * blockDim.x + threadIdx.x;
    size_t stride = (size_t)gridDim.x * blockDim.x;
    for (; i < n; i += stride) p[i] = 0.0f;
}

// Transpose + pad fp32 weight [K_real][N_real] -> f16 [N_pad][K_pad], pad bias
__global__ void convT(const float* __restrict__ W, const float* __restrict__ b,
                      _Float16* __restrict__ Wt, float* __restrict__ bp,
                      int K_real, int N_real, int K_pad, int N_pad) {
    long long t = (long long)blockIdx.x * blockDim.x + threadIdx.x;
    long long total = (long long)N_pad * K_pad;
    if (t >= total) return;
    int n = (int)(t / K_pad);
    int k = (int)(t % K_pad);
    float v = (n < N_real && k < K_real) ? W[(long long)k * N_real + n] : 0.0f;
    Wt[t] = (_Float16)v;
    if (k == 0) bp[n] = (n < N_real) ? b[n] : 0.0f;
}

// Pad-convert fp32 [M][K_real] -> f16 [M][K_pad]
__global__ void pad_convert(const float* __restrict__ src, _Float16* __restrict__ dst,
                            int M, int K_real, int K_pad) {
    long long t = (long long)blockIdx.x * blockDim.x + threadIdx.x;
    long long total = (long long)M * K_pad;
    if (t >= total) return;
    int m = (int)(t / K_pad);
    int k = (int)(t % K_pad);
    dst[t] = (_Float16)((k < K_real) ? src[(long long)m * K_real + k] : 0.0f);
}

// -------------------------------------------------- GINE message + scatter
__global__ void edge_msg(const float* __restrict__ x, const int* __restrict__ ei,
                         const float* __restrict__ ex, const float* __restrict__ We,
                         const float* __restrict__ be, float* __restrict__ agg) {
    int e = blockIdx.x * blockDim.x + threadIdx.x;
    if (e >= N_EDGES) return;
    int s = ei[e];
    int d = ei[N_EDGES + e];
    float ef[EDGE_DIM];
#pragma unroll
    for (int i = 0; i < EDGE_DIM; ++i) ef[i] = ex[(long long)e * EDGE_DIM + i];
    const float* xs = x + (long long)s * IN_DIM;
    float* ag = agg + (long long)d * IN_DIM;
#pragma unroll
    for (int j = 0; j < IN_DIM; ++j) {
        float acc = be[j] + xs[j];
#pragma unroll
        for (int i = 0; i < EDGE_DIM; ++i) acc = fmaf(ef[i], We[i * IN_DIM + j], acc);
        acc = fmaxf(acc, 0.0f);
        atomicAdd(&ag[j], acc);
    }
}

// h0 = (1+eps)*x + agg  -> f16 padded K 28->32
__global__ void node_update(const float* __restrict__ x, const float* __restrict__ agg,
                            const float* __restrict__ eps, _Float16* __restrict__ h0) {
    int t = blockIdx.x * blockDim.x + threadIdx.x;
    if (t >= N_NODES * K1P) return;
    int m = t / K1P, k = t % K1P;
    float v = 0.0f;
    if (k < IN_DIM) {
        long long idx = (long long)m * IN_DIM + k;
        v = (1.0f + eps[0]) * x[idx] + agg[idx];
    }
    h0[t] = (_Float16)v;
}

// ---------------------------------------------------------- WMMA GEMM core
// Build a v16h fragment from LDS: elements 0..7 at off_lo, 8..15 at off_hi.
__device__ __forceinline__ v16h frag_ld(const _Float16* base, int r, int off_lo, int off_hi) {
    const _Float16* p = base + r * LDS_STRIDE;
    v8h lo = *(const v8h*)(p + off_lo);
    v8h hi = *(const v8h*)(p + off_hi);
    v16h t;
#pragma unroll
    for (int i = 0; i < 8; ++i) { t[i] = lo[i]; t[i + 8] = hi[i]; }
    return t;
}

// D = act(A @ Bt^T + bias). A:[M][K_pad] f16 row-major, Bt:[N_pad][K_pad] f16.
// Block tile 128(M) x 64(N); 8 waves (4 along M x 2 along N), 32x32 per wave.
// LDS double-buffered staging via async global->LDS copies.
// act: 0 none, 1 relu, 2 leaky(0.01), 3 sigmoid.
// outH != nullptr -> f16 [M][N_pad] out; else f32 outF[M][ldc] for n < N_real.
__global__ void __launch_bounds__(256)
wmma_gemm(const _Float16* __restrict__ A, const _Float16* __restrict__ Bt,
          const float* __restrict__ bias, float* __restrict__ outF,
          _Float16* __restrict__ outH, int M, int N_pad, int N_real,
          int K_pad, int ldc, int act) {
    __shared__ _Float16 sA[2][128 * LDS_STRIDE];
    __shared__ _Float16 sB[2][64 * LDS_STRIDE];

    const int tid  = threadIdx.x;
    const int lane = tid & 31;
    const int wave = tid >> 5;
    const int wm = wave & 3;            // 4 waves along M
    const int wn = wave >> 2;           // 2 waves along N
    const int m_blk = blockIdx.y * 128;
    const int n_blk = blockIdx.x * 64;
    const int row = lane & 15;
    const int hlf = lane >> 4;

    // per-thread copy assignments (16B chunks; tile row = chunk/4, col = chunk%4)
    const int ar0 = tid >> 2;                 // A rows 0..63   (chunk = tid)
    const int ar1 = (tid + 256) >> 2;         // A rows 64..127 (chunk = tid+256)
    const int aq  = (tid & 3) * 8;
    const int br  = tid >> 2;                 // B rows 0..63
    const int bq  = (tid & 3) * 8;

    auto issue = [&](int s, int k) {
        cp16(A + (size_t)(m_blk + ar0) * K_pad + k + aq, &sA[s][ar0 * LDS_STRIDE + aq]);
        cp16(A + (size_t)(m_blk + ar1) * K_pad + k + aq, &sA[s][ar1 * LDS_STRIDE + aq]);
        cp16(Bt + (size_t)(n_blk + br) * K_pad + k + bq, &sB[s][br * LDS_STRIDE + bq]);
    };

    v8f acc00 = {}, acc01 = {}, acc10 = {}, acc11 = {};

    const int nk = K_pad >> 5;
    issue(0, 0);
    for (int kt = 0; kt < nk; ++kt) {
        wait_cp();
        __syncthreads();
        if (kt + 1 < nk) issue((kt + 1) & 1, (kt + 1) << 5);

        const _Float16* As = sA[kt & 1];
        const _Float16* Bs = sB[kt & 1];
        // A-layout: lanes 0-15 K=0..7/16..23, lanes 16-31 K=8..15/24..31
        v16h a0 = frag_ld(As, wm * 32 + row,      8 * hlf, 16 + 8 * hlf);
        v16h a1 = frag_ld(As, wm * 32 + 16 + row, 8 * hlf, 16 + 8 * hlf);
        // B-layout: lanes 0-15 K=0..7/16..23, lanes 16-31 K=8..15/24..31 for column n=lane&15
        v16h b0 = frag_ld(Bs, wn * 32 + row,      8 * hlf, 16 + 8 * hlf);
        v16h b1 = frag_ld(Bs, wn * 32 + 16 + row, 8 * hlf, 16 + 8 * hlf);

        acc00 = __builtin_amdgcn_wmma_f32_16x16x32_f16(false, a0, false, b0,
                                                       (short)0, acc00, false, false);
        acc01 = __builtin_amdgcn_wmma_f32_16x16x32_f16(false, a0, false, b1,
                                                       (short)0, acc01, false, false);
        acc10 = __builtin_amdgcn_wmma_f32_16x16x32_f16(false, a1, false, b0,
                                                       (short)0, acc10, false, false);
        acc11 = __builtin_amdgcn_wmma_f32_16x16x32_f16(false, a1, false, b1,
                                                       (short)0, acc11, false, false);
    }

    const int m_base = m_blk + wm * 32;
    const int n_base = n_blk + wn * 32;
    v8f accs[2][2] = {{acc00, acc01}, {acc10, acc11}};
#pragma unroll
    for (int tm = 0; tm < 2; ++tm) {
#pragma unroll
        for (int tn = 0; tn < 2; ++tn) {
            int n = n_base + tn * 16 + row;
            float bv = bias[n];
#pragma unroll
            for (int r = 0; r < 8; ++r) {
                int m = m_base + tm * 16 + r + 8 * hlf;
                float v = accs[tm][tn][r] + bv;
                if (act == 1)      v = fmaxf(v, 0.0f);
                else if (act == 2) v = (v > 0.0f) ? v : 0.01f * v;
                else if (act == 3) v = 1.0f / (1.0f + __expf(-v));
                if (outH)            outH[(size_t)m * N_pad + n] = (_Float16)v;
                else if (n < N_real) outF[(size_t)m * ldc + n] = v;
            }
        }
    }
}

// -------------------------------------------- BatchNorm stats (2-stage) ---
__global__ void col_partial(const float* __restrict__ h2, float* __restrict__ sumc,
                            float* __restrict__ sumsq) {
    int c = blockIdx.x * blockDim.x + threadIdx.x;
    if (c >= HID) return;
    int r0 = blockIdx.y * 512;
    float s = 0.0f, sq = 0.0f;
    for (int r = r0; r < r0 + 512; ++r) {
        float v = h2[(size_t)r * HID + c];
        s += v;
        sq += v * v;
    }
    atomicAdd(&sumc[c], s);
    atomicAdd(&sumsq[c], sq);
}

__global__ void finalize_stats(const float* __restrict__ sumc, const float* __restrict__ sumsq,
                               float* __restrict__ mean, float* __restrict__ var) {
    int c = blockIdx.x * blockDim.x + threadIdx.x;
    if (c >= HID) return;
    float m = sumc[c] * (1.0f / N_NODES);
    mean[c] = m;
    var[c] = sumsq[c] * (1.0f / N_NODES) - m * m;
}

// BN applied per element, summed per graph (batch is sorted -> run-length atomics)
__global__ void pool_bn(const float* __restrict__ h2, const int* __restrict__ batch,
                        const float* __restrict__ gamma, const float* __restrict__ beta,
                        const float* __restrict__ mean, const float* __restrict__ var,
                        float* __restrict__ pooled) {
    int c = blockIdx.x * blockDim.x + threadIdx.x;
    if (c >= HID) return;
    int n0 = blockIdx.y * 128;
    float g = gamma[c], bt = beta[c], mu = mean[c];
    float rs = rsqrtf(var[c] + BN_EPS);
    int gcur = -1;
    float accum = 0.0f;
    for (int n = n0; n < n0 + 128; ++n) {
        int gr = batch[n];
        if (gr != gcur) {
            if (gcur >= 0) atomicAdd(&pooled[(size_t)gcur * HID + c], accum);
            gcur = gr;
            accum = 0.0f;
        }
        float v = h2[(size_t)n * HID + c];
        accum += g * (v - mu) * rs + bt;
    }
    if (gcur >= 0) atomicAdd(&pooled[(size_t)gcur * HID + c], accum);
}

// ------------------------------------------------------------------ driver
extern "C" void kernel_launch(void* const* d_in, const int* in_sizes, int n_in,
                              void* d_out, int out_size, void* d_ws, size_t ws_size,
                              hipStream_t stream) {
    const float* x       = (const float*)d_in[0];
    const int*   ei      = (const int*)d_in[1];
    const float* ex      = (const float*)d_in[2];
    const int*   batch   = (const int*)d_in[3];
    const float* eps     = (const float*)d_in[4];
    const float* We      = (const float*)d_in[5];
    const float* be      = (const float*)d_in[6];
    const float* W1      = (const float*)d_in[7];
    const float* b1      = (const float*)d_in[8];
    const float* W2      = (const float*)d_in[9];
    const float* b2      = (const float*)d_in[10];
    const float* gamma   = (const float*)d_in[11];
    const float* beta    = (const float*)d_in[12];
    const float* Wp      = (const float*)d_in[13];
    const float* bp      = (const float*)d_in[14];
    const float* Wm      = (const float*)d_in[15];
    const float* bm      = (const float*)d_in[16];
    float* out = (float*)d_out;

    size_t off = 0;
    auto carve = [&](size_t bytes) -> void* {
        void* p = (char*)d_ws + off;
        off += (bytes + 255) & ~(size_t)255;
        return p;
    };
    float*    agg     = (float*)   carve((size_t)N_NODES * IN_DIM * 4);
    _Float16* h0      = (_Float16*)carve((size_t)N_NODES * K1P * 2);
    _Float16* W1t     = (_Float16*)carve((size_t)HIDP * K1P * 2);
    float*    b1p     = (float*)   carve((size_t)HIDP * 4);
    _Float16* h1      = (_Float16*)carve((size_t)N_NODES * HIDP * 2);
    _Float16* W2t     = (_Float16*)carve((size_t)HIDP * HIDP * 2);
    float*    b2p     = (float*)   carve((size_t)HIDP * 4);
    float*    h2      = (float*)   carve((size_t)N_NODES * HID * 4);
    float*    sumc    = (float*)   carve((size_t)HID * 4);
    float*    sumsq   = (float*)   carve((size_t)HID * 4);
    float*    mean    = (float*)   carve((size_t)HID * 4);
    float*    var     = (float*)   carve((size_t)HID * 4);
    float*    pooled  = (float*)   carve((size_t)NUM_GRAPHS * HID * 4);
    _Float16* pooledh = (_Float16*)carve((size_t)NUM_GRAPHS * HIDP * 2);
    _Float16* Wpt     = (_Float16*)carve((size_t)PROJP * HIDP * 2);
    float*    bpp     = (float*)   carve((size_t)PROJP * 4);
    _Float16* qh      = (_Float16*)carve((size_t)NUM_GRAPHS * PROJP * 2);
    _Float16* Wmt     = (_Float16*)carve((size_t)OUT_DIM * PROJP * 2);
    float*    bmp     = (float*)   carve((size_t)OUT_DIM * 4);

    // zero accumulators (d_ws is not re-poisoned between replays)
    zero_f32<<<2048, 256, 0, stream>>>(agg, (size_t)N_NODES * IN_DIM);
    zero_f32<<<64,   256, 0, stream>>>(sumc, (size_t)HID);
    zero_f32<<<64,   256, 0, stream>>>(sumsq, (size_t)HID);
    zero_f32<<<2048, 256, 0, stream>>>(pooled, (size_t)NUM_GRAPHS * HID);

    // weight transposition + padding to f16
    {
        long long t1 = (long long)HIDP * K1P;
        convT<<<(int)((t1 + 255) / 256), 256, 0, stream>>>(W1, b1, W1t, b1p, IN_DIM, HID, K1P, HIDP);
        long long t2 = (long long)HIDP * HIDP;
        convT<<<(int)((t2 + 255) / 256), 256, 0, stream>>>(W2, b2, W2t, b2p, HID, HID, HIDP, HIDP);
        long long tp = (long long)PROJP * HIDP;
        convT<<<(int)((tp + 255) / 256), 256, 0, stream>>>(Wp, bp, Wpt, bpp, HID, PROJ, HIDP, PROJP);
        long long tm = (long long)OUT_DIM * PROJP;
        convT<<<(int)((tm + 255) / 256), 256, 0, stream>>>(Wm, bm, Wmt, bmp, PROJ, OUT_DIM, PROJP, OUT_DIM);
    }

    // GINE message + scatter-add
    edge_msg<<<(N_EDGES + 255) / 256, 256, 0, stream>>>(x, ei, ex, We, be, agg);
    node_update<<<(N_NODES * K1P + 255) / 256, 256, 0, stream>>>(x, agg, eps, h0);

    // GEMM1: [32768,32] x [32,3008] -> h1 f16 (bias + relu)
    {
        dim3 grid(HIDP / 64, N_NODES / 128);
        wmma_gemm<<<grid, 256, 0, stream>>>(h0, W1t, b1p, nullptr, h1,
                                            N_NODES, HIDP, HID, K1P, 0, 1);
    }
    // GEMM2: [32768,3008] x [3008,3008] -> h2 f32 (bias + relu)  — dominant
    {
        dim3 grid(HIDP / 64, N_NODES / 128);
        wmma_gemm<<<grid, 256, 0, stream>>>(h1, W2t, b2p, h2, nullptr,
                                            N_NODES, HIDP, HID, HIDP, HID, 1);
    }

    // BatchNorm stats + fused BN/global_add_pool
    {
        dim3 grid((HID + 255) / 256, N_NODES / 512);
        col_partial<<<grid, 256, 0, stream>>>(h2, sumc, sumsq);
        finalize_stats<<<(HID + 255) / 256, 256, 0, stream>>>(sumc, sumsq, mean, var);
        dim3 grid2((HID + 255) / 256, N_NODES / 128);
        pool_bn<<<grid2, 256, 0, stream>>>(h2, batch, gamma, beta, mean, var, pooled);
    }
    pad_convert<<<(int)(((long long)NUM_GRAPHS * HIDP + 255) / 256), 256, 0, stream>>>(
        pooled, pooledh, NUM_GRAPHS, HID, HIDP);

    // GEMM-P: [256,3008] x [3008,3776] -> q f16 (bias + leaky_relu)
    {
        dim3 grid(PROJP / 64, NUM_GRAPHS / 128);
        wmma_gemm<<<grid, 256, 0, stream>>>(pooledh, Wpt, bpp, nullptr, qh,
                                            NUM_GRAPHS, PROJP, PROJ, HIDP, 0, 2);
    }
    // GEMM-M: [256,3776] x [3776,128] -> out f32 (bias + sigmoid)
    {
        dim3 grid(OUT_DIM / 64, NUM_GRAPHS / 128);
        wmma_gemm<<<grid, 256, 0, stream>>>(qh, Wmt, bmp, out, nullptr,
                                            NUM_GRAPHS, OUT_DIM, OUT_DIM, PROJP, OUT_DIM, 3);
    }
}
